// LogisticShapeletsLearner_36842229465233
// MI455X (gfx1250) — matrix operand order repaired and checked by reading, not compiled, not run on hardware
//
#include <hip/hip_runtime.h>
#include <hip/hip_bf16.h>

typedef __attribute__((ext_vector_type(16))) _Float16 v16h;
typedef __attribute__((ext_vector_type(8)))  float    v8f;

#define SM_ALPHA (-30.0f)
#define SM_EPS   (1e-4f)

static constexpr int N_SERIES  = 512;
static constexpr int T_LEN     = 2048;
static constexpr int K_SHP     = 64;
static constexpr int PS        = 2064;   // prefix-sum row stride (floats), >= T_LEN+1
static constexpr int TP        = 2176;   // zero-padded series length staged in LDS
static constexpr int N_FEAT    = 192;
static constexpr int N_CLASSES = 10;

// ---------------------------------------------------------------------------
// Kernel 1: per-series exclusive prefix sum of squares (fp32).
// P[n][t] = sum_{u<t} series[n][u]^2,  t in [0, T_LEN], so that
// sum(x_win^2) for window w of length L = P[w+L] - P[w].
// ---------------------------------------------------------------------------
__global__ __launch_bounds__(256) void prefix_sq_kernel(const float* __restrict__ series,
                                                        float* __restrict__ P) {
  const int n = blockIdx.x;
  const int tid = threadIdx.x;
  __shared__ float part[256];
  const float* row = series + (size_t)n * T_LEN;
  float xs[8];
  float s = 0.f;
#pragma unroll
  for (int j = 0; j < 8; ++j) {
    float v = row[tid * 8 + j];
    xs[j] = v * v;
    s += xs[j];
  }
  part[tid] = s;
  __syncthreads();
  if (tid == 0) {
    float run = 0.f;
    for (int i = 0; i < 256; ++i) { float t = part[i]; part[i] = run; run += t; }
    P[(size_t)n * PS + T_LEN] = run;
  }
  __syncthreads();
  float base = part[tid];
  float* Pr = P + (size_t)n * PS;
#pragma unroll
  for (int j = 0; j < 8; ++j) {
    Pr[tid * 8 + j] = base;
    base += xs[j];
  }
}

// ---------------------------------------------------------------------------
// Kernel 2 (x3 scales): WMMA sliding correlation + softmin pooling.
// Grid: (K_SHP/16, N_SERIES). Block: 256 threads = 8 waves.
// Each block handles one (series n, 16-shapelet group kg); waves stripe over
// tiles of 16 windows. D = A(16x32 shapelets, f16) x B(32x16 windows, f16)
// accumulated over CHUNKS K-chunks via v_wmma_f32_16x16x32_f16.
// ---------------------------------------------------------------------------
template <int CHUNKS>
__global__ __launch_bounds__(256) void softmin_kernel(const float* __restrict__ series,
                                                      const float* __restrict__ shp,
                                                      const float* __restrict__ P,
                                                      float* __restrict__ feat,
                                                      int featOff) {
  constexpr int L  = 32 * CHUNKS;
  constexpr int W  = T_LEN - L + 1;
  constexpr int NT = (W + 15) / 16;

  const int kg   = blockIdx.x;
  const int n    = blockIdx.y;
  const int tid  = threadIdx.x;
  const int wid  = tid >> 5;
  const int lane = tid & 31;
  const int wloc = lane & 15;   // window-in-tile (B column, D column)
  const int hi   = lane >> 4;   // lane half selects K sub-range / M sub-range

  __shared__ _Float16 sS[TP];        // full padded series, f16
  __shared__ _Float16 sA[16 * L];    // shapelet group, f16, row-major [j][L]
  __shared__ float    s2loc[16];     // sum(shp^2) per shapelet
  __shared__ float    redE[8][16];
  __shared__ float    redDE[8][16];

  // --- stage series (convert f32 -> f16, zero pad tail) ---
  const float* srow = series + (size_t)n * T_LEN;
  for (int t = tid; t < TP; t += 256)
    sS[t] = (t < T_LEN) ? (_Float16)srow[t] : (_Float16)0.f;
  // --- stage shapelet group ---
  for (int idx = tid; idx < 16 * L; idx += 256) {
    int j = idx / L, p = idx % L;
    sA[idx] = (_Float16)shp[(size_t)(kg * 16 + j) * L + p];
  }
  // --- sum of squares per shapelet (fp32, exact) ---
  if (tid < 16) {
    const float* sp = shp + (size_t)(kg * 16 + tid) * L;
    float s = 0.f;
    for (int p = 0; p < L; ++p) { float v = sp[p]; s += v * v; }
    s2loc[tid] = s;
  }
  __syncthreads();

  // A fragments (ISA 16-bit A 16x32 layout): row M = wloc,
  // slot h -> K = h + (h>=8 ? 8 : 0) + 8*hi. Loaded once, reused every tile.
  v16h a[CHUNKS];
#pragma unroll
  for (int c = 0; c < CHUNKS; ++c) {
#pragma unroll
    for (int h = 0; h < 16; ++h) {
      const int kk = h + (h >= 8 ? 8 : 0) + 8 * hi;
      a[c][h] = sA[wloc * L + c * 32 + kk];
    }
  }

  // Hoist per-lane shapelet sum-of-squares into registers (loop invariant).
  // D layout: VGPR r -> shapelet M = r + 8*hi.
  float s2r[8];
#pragma unroll
  for (int r = 0; r < 8; ++r) s2r[r] = s2loc[r + 8 * hi];

  float accE[8], accDE[8];
#pragma unroll
  for (int r = 0; r < 8; ++r) { accE[r] = 0.f; accDE[r] = 0.f; }

  const float* Pr   = P + (size_t)n * PS;
  const float  invL = 1.0f / (float)L;
  const float  aScl = SM_ALPHA * invL;   // exp argument = raw * (ALPHA/L)

  for (int tile = wid; tile < NT; tile += 8) {
    const int base = tile * 16;
    v8f acc = {};
#pragma unroll
    for (int c = 0; c < CHUNKS; ++c) {
      // B fragment (ISA 16-bit B 32x16 layout): column = wloc,
      // slot h -> K = h + (h>=8 ? 8 : 0) + 8*hi; element (k, w) = series[base + w + c*32 + k]
      v16h b;
      const int off0 = base + wloc + c * 32 + 8 * hi;
#pragma unroll
      for (int h = 0; h < 16; ++h) b[h] = sS[off0 + h + (h >= 8 ? 8 : 0)];
      acc = __builtin_amdgcn_wmma_f32_16x16x32_f16(false, a[c], false, b,
                                                   (short)0, acc, false, false);
    }
    const int   w     = base + wloc;
    const float vmask = (w < W) ? 1.0f : 0.0f;      // branchless tail mask
    const int   wc    = (w < W) ? w : (W - 1);
    const float sumx2 = Pr[wc + L] - Pr[wc];
#pragma unroll
    for (int r = 0; r < 8; ++r) {
      const float raw = sumx2 - 2.0f * acc[r] + s2r[r];  // L * d
      const float d   = raw * invL;
      const float e   = (__expf(raw * aScl) + SM_EPS) * vmask;
      accE[r]  += e;
      accDE[r] += d * e;
    }
  }

  // reduce over the 16 windows-per-tile lanes (xor masks <= 8 stay in half)
#pragma unroll
  for (int r = 0; r < 8; ++r) {
    float e = accE[r], de = accDE[r];
#pragma unroll
    for (int off = 8; off >= 1; off >>= 1) {
      e  += __shfl_xor(e,  off, 32);
      de += __shfl_xor(de, off, 32);
    }
    if (wloc == 0) { redE[wid][r + 8 * hi] = e; redDE[wid][r + 8 * hi] = de; }
  }
  __syncthreads();
  if (tid < 16) {
    float E = 0.f, DE = 0.f;
#pragma unroll
    for (int v = 0; v < 8; ++v) { E += redE[v][tid]; DE += redDE[v][tid]; }
    feat[(size_t)n * N_FEAT + featOff + kg * 16 + tid] = DE / E;
  }
}

// ---------------------------------------------------------------------------
// Kernel 3: linear head + softmax. One thread per series row (trivial cost).
// ---------------------------------------------------------------------------
__global__ __launch_bounds__(256) void head_kernel(const float* __restrict__ feat,
                                                   const float* __restrict__ Wm,
                                                   const float* __restrict__ bias,
                                                   float* __restrict__ out) {
  const int n = blockIdx.x * blockDim.x + threadIdx.x;
  if (n >= N_SERIES) return;
  float lg[N_CLASSES];
#pragma unroll
  for (int c = 0; c < N_CLASSES; ++c) lg[c] = bias[c];
  const float* fr = feat + (size_t)n * N_FEAT;
  for (int f = 0; f < N_FEAT; ++f) {
    const float v = fr[f];
#pragma unroll
    for (int c = 0; c < N_CLASSES; ++c) lg[c] += v * Wm[f * N_CLASSES + c];
  }
  float m = lg[0];
#pragma unroll
  for (int c = 1; c < N_CLASSES; ++c) m = fmaxf(m, lg[c]);
  float sum = 0.f;
#pragma unroll
  for (int c = 0; c < N_CLASSES; ++c) { lg[c] = __expf(lg[c] - m); sum += lg[c]; }
  const float inv = 1.0f / sum;
#pragma unroll
  for (int c = 0; c < N_CLASSES; ++c) out[(size_t)n * N_CLASSES + c] = lg[c] * inv;
}

// ---------------------------------------------------------------------------
extern "C" void kernel_launch(void* const* d_in, const int* in_sizes, int n_in,
                              void* d_out, int out_size, void* d_ws, size_t ws_size,
                              hipStream_t stream) {
  (void)in_sizes; (void)n_in; (void)out_size; (void)ws_size;
  const float* series = (const float*)d_in[0];
  const float* shp1   = (const float*)d_in[1];
  const float* shp2   = (const float*)d_in[2];
  const float* shp3   = (const float*)d_in[3];
  const float* Wm     = (const float*)d_in[4];
  const float* bias   = (const float*)d_in[5];
  float* out = (float*)d_out;

  float* P    = (float*)d_ws;                 // [N_SERIES][PS] prefix sums
  float* feat = P + (size_t)N_SERIES * PS;    // [N_SERIES][N_FEAT]

  prefix_sq_kernel<<<N_SERIES, 256, 0, stream>>>(series, P);

  dim3 grid(K_SHP / 16, N_SERIES);
  softmin_kernel<1><<<grid, 256, 0, stream>>>(series, shp1, P, feat, 0);
  softmin_kernel<2><<<grid, 256, 0, stream>>>(series, shp2, P, feat, 64);
  softmin_kernel<3><<<grid, 256, 0, stream>>>(series, shp3, P, feat, 128);

  head_kernel<<<(N_SERIES + 255) / 256, 256, 0, stream>>>(feat, Wm, bias, out);
}
